// AutoEncoderTopK_10376640987240
// MI455X (gfx1250) — compile-verified, matrix operand order checked by reading, not run on hardware
//
#include <hip/hip_runtime.h>
#include <hip/hip_bf16.h>
#include <stdint.h>

// ---------------------------------------------------------------------------
// TopK sparse autoencoder forward for MI455X (gfx1250, wave32, WMMA).
//   pass 0a: xc_bf16  = bf16(x - b_dec)                    [B][D_IN]
//   pass 0b: Wt_bf16  = bf16(W_enc^T)                      [D_SAE][D_IN]
//   pass 1 : acts     = relu(xc @ W_enc + b_enc)  (WMMA, async-LDS staged)
//   pass 2 : per-row exact top-64 via 32-bit radix select
//   pass 3 : x_hat    = b_dec + sum_j val_j * W_dec[idx_j] [B][D_IN]
// Workspace requirement: ~709 MB.
// ---------------------------------------------------------------------------

#define B_ROWS 4096
#define D_IN   2304
#define D_SAE  32768
#define TOPK   64

typedef __attribute__((ext_vector_type(16))) __bf16 v16bf;
typedef __attribute__((ext_vector_type(8)))  float  v8f;
typedef __attribute__((ext_vector_type(4)))  int    v4i;

union FragU { uint4 q[2]; v16bf v; };

// ---- CDNA5 async global->LDS support (guarded; fallback = reg pipeline) ----
#if defined(__has_builtin)
#  if __has_builtin(__builtin_amdgcn_global_load_async_to_lds_b128)
#    define ASYNC_LDS 1
#  endif
#endif
#ifndef ASYNC_LDS
#  define ASYNC_LDS 0
#endif

#if defined(__has_builtin)
#  if __has_builtin(__builtin_amdgcn_s_wait_asynccnt)
#    define WAIT_ASYNC(n) __builtin_amdgcn_s_wait_asynccnt(n)
#  endif
#endif
#ifndef WAIT_ASYNC
#  define WAIT_ASYNC(n) asm volatile("s_wait_asynccnt %0" :: "i"(n) : "memory")
#endif

#if ASYNC_LDS
// Signature (from clang diagnostic): param0 = v4i addrspace(1)*, param1 =
// v4i addrspace(3)*, then (int offset, int cpol).
typedef __attribute__((address_space(1))) v4i gv4i_t;
typedef __attribute__((address_space(3))) v4i sv4i_t;
__device__ __forceinline__ void async_cp16(const unsigned short* g, unsigned short* l) {
    __builtin_amdgcn_global_load_async_to_lds_b128(
        (gv4i_t*)(void*)g,
        (sv4i_t*)(void*)l,
        0, 0);
}
#endif

__device__ __forceinline__ unsigned short f2bf(float f) {
    unsigned int u = __float_as_uint(f);
    unsigned int r = u + 0x7FFFu + ((u >> 16) & 1u);   // round-to-nearest-even
    return (unsigned short)(r >> 16);
}

// ---------------------------------------------------------------------------
// Pass 0a: xc = bf16(x - b_dec), elementwise over B*D_IN
// ---------------------------------------------------------------------------
__global__ __launch_bounds__(256) void k_center_cast(
    const float* __restrict__ x, const float* __restrict__ b_dec,
    unsigned short* __restrict__ xc)
{
    int i = blockIdx.x * 256 + threadIdx.x;
    if (i < B_ROWS * D_IN) {
        int col = i % D_IN;
        xc[i] = f2bf(x[i] - b_dec[col]);
    }
}

// ---------------------------------------------------------------------------
// Pass 0b: Wt[n][k] = bf16(W_enc[k][n]); 32x32 LDS tile transpose
// grid: (D_SAE/32, D_IN/32), block 256
// ---------------------------------------------------------------------------
__global__ __launch_bounds__(256) void k_transpose_cast(
    const float* __restrict__ W, unsigned short* __restrict__ Wt)
{
    __shared__ unsigned short tile[32][33];
    const int n0 = blockIdx.x * 32;
    const int k0 = blockIdx.y * 32;
    const int tx = threadIdx.x & 31;
    const int ty = threadIdx.x >> 5;
#pragma unroll
    for (int r = ty; r < 32; r += 8)
        tile[r][tx] = f2bf(W[(size_t)(k0 + r) * D_SAE + n0 + tx]);  // tile[k][n]
    __syncthreads();
#pragma unroll
    for (int r = ty; r < 32; r += 8)
        Wt[(size_t)(n0 + r) * D_IN + k0 + tx] = tile[tx][r];        // Wt[n][k]
}

// ---------------------------------------------------------------------------
// Pass 1: GEMM via v_wmma_f32_16x16x32_bf16.
// WG tile 128(M) x 256(N); 8 waves (2x4); wave tile 64x64 = 4x4 WMMA tiles.
// Register budget: 128 acc + 32 A-frags held + 8 B-frag streamed (~170 live)
// to stay inside the 256-VGPR direct window (avoid s_set_vgpr_msb shuffling).
// Staging: double-buffered async global->LDS, 1 barrier / K-step.
// ---------------------------------------------------------------------------
#define GT_M 128
#define GT_N 256
#define GT_K 32

__device__ __forceinline__ void tile_mma(
    const unsigned short (&A)[GT_M][GT_K],
    const unsigned short (&B)[GT_N][GT_K],
    int waveM, int waveN, int lane, v8f (&acc)[4][4])
{
    const int kb   = (lane < 16) ? 0 : 8;
    const int rSub = lane & 15;
    // Hold all 4 A fragments; stream B fragments one at a time.
    FragU af[4];
#pragma unroll
    for (int i = 0; i < 4; ++i) {
        int m = waveM * 64 + i * 16 + rSub;
        af[i].q[0] = *(const uint4*)&A[m][kb];
        af[i].q[1] = *(const uint4*)&A[m][kb + 16];
    }
#pragma unroll
    for (int j = 0; j < 4; ++j) {
        int n = waveN * 64 + j * 16 + rSub;
        FragU bf;
        bf.q[0] = *(const uint4*)&B[n][kb];
        bf.q[1] = *(const uint4*)&B[n][kb + 16];
#pragma unroll
        for (int i = 0; i < 4; ++i)
            acc[i][j] = __builtin_amdgcn_wmma_f32_16x16x32_bf16(
                false, af[i].v, false, bf.v, (short)0, acc[i][j], false, false);
    }
}

__global__ __launch_bounds__(256) void k_enc_gemm(
    const unsigned short* __restrict__ xc,   // [B][D_IN]     bf16
    const unsigned short* __restrict__ wt,   // [D_SAE][D_IN] bf16
    const float* __restrict__ b_enc,
    float* __restrict__ acts)                // [B][D_SAE]    f32
{
#if ASYNC_LDS
    __shared__ __align__(16) unsigned short As[2][GT_M][GT_K];  // 16 KB
    __shared__ __align__(16) unsigned short Bs[2][GT_N][GT_K];  // 32 KB
#else
    __shared__ __align__(16) unsigned short As[1][GT_M][GT_K];
    __shared__ __align__(16) unsigned short Bs[1][GT_N][GT_K];
#endif

    const int tid   = threadIdx.x;
    const int lane  = tid & 31;
    const int wave  = tid >> 5;
    const int waveM = wave >> 2;             // 0..1
    const int waveN = wave & 3;              // 0..3
    const int blockM = blockIdx.y * GT_M;
    const int blockN = blockIdx.x * GT_N;

    // Straight-line staging coordinates: thread owns fixed (row, 16B chunk)
    // slots; only the K offset varies per step.
    const int cA  = (tid & 3) * 8;           // bf16 column within 32-wide tile
    const int rA0 = tid >> 2;                // 0..63
    const int rA1 = rA0 + 64;                // 64..127
    const unsigned short* gA0 = xc + (size_t)(blockM + rA0) * D_IN + cA;
    const unsigned short* gA1 = xc + (size_t)(blockM + rA1) * D_IN + cA;
    const unsigned short* gB0 = wt + (size_t)(blockN + rA0) * D_IN + cA;
    const unsigned short* gB1 = gB0 + (size_t)64  * D_IN;
    const unsigned short* gB2 = gB0 + (size_t)128 * D_IN;
    const unsigned short* gB3 = gB0 + (size_t)192 * D_IN;

    v8f acc[4][4] = {};
    const int NT = D_IN / GT_K;              // 72 K-steps

#if ASYNC_LDS
    auto stage = [&](int buf, int kk) {
        async_cp16(gA0 + kk, &As[buf][rA0][cA]);
        async_cp16(gA1 + kk, &As[buf][rA1][cA]);
        async_cp16(gB0 + kk, &Bs[buf][rA0      ][cA]);
        async_cp16(gB1 + kk, &Bs[buf][rA0 +  64][cA]);
        async_cp16(gB2 + kk, &Bs[buf][rA0 + 128][cA]);
        async_cp16(gB3 + kk, &Bs[buf][rA0 + 192][cA]);
    };
    stage(0, 0);
    int cur = 0;
#pragma unroll 1   // keep a single fragment set live; no double-unroll
    for (int kt = 0; kt < NT; ++kt) {
        WAIT_ASYNC(0);            // this wave's writes for tile kt landed
        __syncthreads();          // ... and everyone else's
        if (kt + 1 < NT) stage(cur ^ 1, (kt + 1) * GT_K);   // overlap w/ MMA
        tile_mma(As[cur], Bs[cur], waveM, waveN, lane, acc);
        cur ^= 1;
        // No trailing barrier needed: stage(kt+2) is issued only after the
        // top barrier of iter kt+1, by which point every wave has finished
        // reading buffer (kt&1) in iter kt.
    }
#else
    // Fallback: register-pipelined synchronous staging.
    uint4 rA[2], rB[4];
    rA[0] = *(const uint4*)gA0;  rA[1] = *(const uint4*)gA1;
    rB[0] = *(const uint4*)gB0;  rB[1] = *(const uint4*)gB1;
    rB[2] = *(const uint4*)gB2;  rB[3] = *(const uint4*)gB3;
#pragma unroll 1
    for (int kt = 0; kt < NT; ++kt) {
        *(uint4*)&As[0][rA0][cA]       = rA[0];
        *(uint4*)&As[0][rA1][cA]       = rA[1];
        *(uint4*)&Bs[0][rA0      ][cA] = rB[0];
        *(uint4*)&Bs[0][rA0 +  64][cA] = rB[1];
        *(uint4*)&Bs[0][rA0 + 128][cA] = rB[2];
        *(uint4*)&Bs[0][rA0 + 192][cA] = rB[3];
        __syncthreads();
        if (kt + 1 < NT) {
            const int kk = (kt + 1) * GT_K;
            rA[0] = *(const uint4*)(gA0 + kk);  rA[1] = *(const uint4*)(gA1 + kk);
            rB[0] = *(const uint4*)(gB0 + kk);  rB[1] = *(const uint4*)(gB1 + kk);
            rB[2] = *(const uint4*)(gB2 + kk);  rB[3] = *(const uint4*)(gB3 + kk);
        }
        tile_mma(As[0], Bs[0], waveM, waveN, lane, acc);
        __syncthreads();
    }
#endif

    // Epilogue: + b_enc, ReLU, store. C layout: VGPR v -> row v (lanes 0-15)
    // or row v+8 (lanes 16-31); N = lane&15.
#pragma unroll
    for (int j = 0; j < 4; ++j) {
        int col   = blockN + waveN * 64 + j * 16 + (lane & 15);
        float bia = b_enc[col];
#pragma unroll
        for (int i = 0; i < 4; ++i) {
            int rowBase = blockM + waveM * 64 + i * 16 + ((lane >> 4) << 3);
#pragma unroll
            for (int v = 0; v < 8; ++v) {
                float val = acc[i][j][v] + bia;
                val = fmaxf(val, 0.0f);
                acts[(size_t)(rowBase + v) * D_SAE + col] = val;
            }
        }
    }
}

// ---------------------------------------------------------------------------
// Pass 2: exact top-64 per row. Post-ReLU values are >= 0, so float order ==
// uint-bit order. 4x 8-bit MSB-first radix passes find the exact 64th-largest
// bit pattern T; then compact (>T) and fill remaining slots from (==T).
// grid: B rows, block 256
// ---------------------------------------------------------------------------
__global__ __launch_bounds__(256) void k_topk(
    const float* __restrict__ acts,
    float* __restrict__ topv, int* __restrict__ topi)
{
    __shared__ unsigned int hist[256];
    __shared__ unsigned int s_sel, s_kneed, s_cntG, s_cntE;
    __shared__ float outV[TOPK];
    __shared__ int   outI[TOPK];

    const int    row = blockIdx.x;
    const int    tid = threadIdx.x;
    const float* rp  = acts + (size_t)row * D_SAE;

    unsigned int prefix = 0, prefmask = 0;
    if (tid == 0) s_kneed = TOPK;
    __syncthreads();

    for (int pass = 0; pass < 4; ++pass) {
        const int shift = 24 - 8 * pass;
        hist[tid] = 0;
        __syncthreads();
        for (int i = tid; i < D_SAE; i += 256) {
            unsigned int key = __float_as_uint(rp[i]);
            if ((key & prefmask) == prefix)
                atomicAdd(&hist[(key >> shift) & 255u], 1u);
        }
        __syncthreads();
        if (tid == 0) {
            unsigned int cum = 0, kneed = s_kneed, sel = 0;
            for (int b = 255; b >= 0; --b) {
                unsigned int c = hist[b];
                if (cum + c >= kneed) { sel = (unsigned)b; s_kneed = kneed - cum; break; }
                cum += c;
            }
            s_sel = sel;
        }
        __syncthreads();
        prefix   |= s_sel << shift;
        prefmask |= 0xFFu  << shift;
        __syncthreads();
    }

    const unsigned int T = prefix;     // exact bit pattern of k-th largest
    if (tid == 0) { s_cntG = 0; s_cntE = 0; }
    __syncthreads();

    for (int i = tid; i < D_SAE; i += 256) {
        unsigned int key = __float_as_uint(rp[i]);
        if (key > T) {
            unsigned int p = atomicAdd(&s_cntG, 1u);   // guaranteed < TOPK
            outV[p] = __uint_as_float(key);
            outI[p] = i;
        }
    }
    __syncthreads();
    const unsigned int base = s_cntG;
    for (int i = tid; i < D_SAE; i += 256) {
        unsigned int key = __float_as_uint(rp[i]);
        if (key == T) {
            unsigned int p = atomicAdd(&s_cntE, 1u);
            if (base + p < TOPK) {
                outV[base + p] = __uint_as_float(key);
                outI[base + p] = i;
            }
        }
    }
    __syncthreads();
    if (tid < TOPK) {
        topv[row * TOPK + tid] = outV[tid];
        topi[row * TOPK + tid] = outI[tid];
    }
}

// ---------------------------------------------------------------------------
// Pass 3: x_hat[row] = b_dec + sum_j val_j * W_dec[idx_j, :]
// grid: B rows, block 256, 9 columns per thread (2304 = 9*256)
// ---------------------------------------------------------------------------
__global__ __launch_bounds__(256) void k_decode(
    const float* __restrict__ topv, const int* __restrict__ topi,
    const float* __restrict__ W_dec, const float* __restrict__ b_dec,
    float* __restrict__ out)
{
    __shared__ float sv[TOPK];
    __shared__ int   si[TOPK];
    const int row = blockIdx.x;
    const int tid = threadIdx.x;
    if (tid < TOPK) {
        sv[tid] = topv[row * TOPK + tid];
        si[tid] = topi[row * TOPK + tid];
    }
    __syncthreads();

    float acc[9];
#pragma unroll
    for (int q = 0; q < 9; ++q) acc[q] = b_dec[tid + q * 256];

    for (int j = 0; j < TOPK; ++j) {
        if (j + 1 < TOPK)   // hint next gathered row toward L2 (global_prefetch_b8)
            __builtin_prefetch(W_dec + (size_t)si[j + 1] * D_IN + tid, 0, 1);
        float w = sv[j];
        if (w != 0.0f) {
            const float* wr = W_dec + (size_t)si[j] * D_IN;
#pragma unroll
            for (int q = 0; q < 9; ++q) acc[q] += w * wr[tid + q * 256];
        }
    }
#pragma unroll
    for (int q = 0; q < 9; ++q)
        out[(size_t)row * D_IN + tid + q * 256] = acc[q];
}

// ---------------------------------------------------------------------------
// Launch
// ---------------------------------------------------------------------------
extern "C" void kernel_launch(void* const* d_in, const int* in_sizes, int n_in,
                              void* d_out, int out_size, void* d_ws, size_t ws_size,
                              hipStream_t stream) {
    const float* x     = (const float*)d_in[0];
    const float* W_enc = (const float*)d_in[1];
    const float* W_dec = (const float*)d_in[2];
    const float* b_enc = (const float*)d_in[3];
    const float* b_dec = (const float*)d_in[4];
    float* out = (float*)d_out;

    char* ws = (char*)d_ws;
    const size_t off_xc   = 0;                                           // 18.9 MB
    const size_t off_wt   = off_xc   + (size_t)B_ROWS * D_IN * 2;        // 151 MB
    const size_t off_acts = off_wt   + (size_t)D_SAE * D_IN * 2;         // 512 MB
    const size_t off_topv = off_acts + (size_t)B_ROWS * D_SAE * 4;       // 1 MB
    const size_t off_topi = off_topv + (size_t)B_ROWS * TOPK * 4;        // 1 MB

    unsigned short* xc   = (unsigned short*)(ws + off_xc);
    unsigned short* wt   = (unsigned short*)(ws + off_wt);
    float*          acts = (float*)(ws + off_acts);
    float*          topv = (float*)(ws + off_topv);
    int*            topi = (int*)(ws + off_topi);

    k_center_cast<<<(B_ROWS * D_IN + 255) / 256, 256, 0, stream>>>(x, b_dec, xc);
    k_transpose_cast<<<dim3(D_SAE / 32, D_IN / 32), 256, 0, stream>>>(W_enc, wt);
    k_enc_gemm<<<dim3(D_SAE / GT_N, B_ROWS / GT_M), 256, 0, stream>>>(xc, wt, b_enc, acts);
    k_topk<<<B_ROWS, 256, 0, stream>>>(acts, topv, topi);
    k_decode<<<B_ROWS, 256, 0, stream>>>(topv, topi, W_dec, b_dec, out);
}
